// WindowAttention_7516192768244
// MI455X (gfx1250) — matrix operand order repaired and checked by reading, not compiled, not run on hardware
//
#include <hip/hip_runtime.h>

// ---------------------------------------------------------------------------
// MI455X (gfx1250) window-attention. bf16 WMMA (v_wmma_f32_16x16x32_bf16)
// for all three GEMM stages; f32 accumulation + f32 softmax for accuracy.
// ---------------------------------------------------------------------------

typedef __attribute__((ext_vector_type(16))) __bf16 v16bf;
typedef __attribute__((ext_vector_type(8)))  float  v8f;
typedef __attribute__((ext_vector_type(4)))  unsigned int u32x4;

union FragBF {
    v16bf v;
    u32x4 u[2];
};

__device__ __host__ inline __bf16 f2bf(float f) {
    union { float f; unsigned u; } c; c.f = f;
    unsigned r = c.u + 0x7FFFu + ((c.u >> 16) & 1u);   // round-to-nearest-even
    unsigned short h = (unsigned short)(r >> 16);
    union { unsigned short s; __bf16 b; } o; o.s = h;
    return o.b;
}

// ---------------------------- constants ------------------------------------
#define BW      512          // number of windows * batch
#define NTOK    98           // tokens per window
#define HEADS   16
#define HD      32           // head dim (== WMMA K!)
#define DIM     512
#define DIM3    1536
#define NW      64           // mask windows
#define NN      (NTOK*NTOK)  // 9604
#define ATT_SCALE 0.17677669529663687f  // 32^-0.5

// ---------------------------------------------------------------------------
// f32 -> bf16 elementwise conversion
// ---------------------------------------------------------------------------
__global__ void f32_to_bf16_kernel(const float* __restrict__ in,
                                   __bf16* __restrict__ out, int n) {
    int i = blockIdx.x * blockDim.x + threadIdx.x;
    if (i < n) out[i] = f2bf(in[i]);
}

// ---------------------------------------------------------------------------
// Gather relative-position bias: biasPre[h][i][j] = table[rel[i*98+j]*16 + h]
// ---------------------------------------------------------------------------
__global__ void bias_gather_kernel(const float* __restrict__ table,
                                   const int* __restrict__ relidx,
                                   float* __restrict__ biasPre) {
    int i = blockIdx.x * blockDim.x + threadIdx.x;      // over NN*HEADS
    if (i >= NN * HEADS) return;
    int h = i & 15;
    int ij = i >> 4;
    biasPre[h * NN + ij] = table[relidx[ij] * HEADS + h];
}

// ---------------------------------------------------------------------------
// Generic bf16 WMMA GEMM:  C[M,N] = A[M,K] * W[N,K]^T + bias[N]
// One wave owns a 16(M) x 64(N) output tile -> 4 accumulators, A reused 4x.
// ---------------------------------------------------------------------------
__device__ inline void storeOut(__bf16* p, float v) { *p = f2bf(v); }
__device__ inline void storeOut(float*  p, float v) { *p = v; }

template <typename OutT>
__global__ __launch_bounds__(256)
void gemm_bf16_wmma(const __bf16* __restrict__ A, const __bf16* __restrict__ W,
                    const float* __restrict__ bias, OutT* __restrict__ C,
                    int M, int N, int K) {
    const int lane = threadIdx.x & 31;
    const int wave = threadIdx.x >> 5;
    const int ngroups = N >> 6;                    // 64-wide N groups
    const int wid = blockIdx.x * 8 + wave;
    if (wid >= (M >> 4) * ngroups) return;         // wave-uniform
    const int mt = wid / ngroups;
    const int ng = wid % ngroups;
    const int ln = lane & 15;
    const int lg = lane >> 4;

    const __bf16* arow = A + (size_t)(mt * 16 + ln) * K;   // A: lane = M row
    const int koffA = lg * 16;                             // A K-chunk select

    v8f acc[4] = {};
#pragma unroll 2
    for (int k0 = 0; k0 < K; k0 += 32) {
        FragBF a;
        a.u[0] = *(const u32x4*)(arow + k0 + koffA);
        a.u[1] = *(const u32x4*)(arow + k0 + koffA + 8);
        __builtin_prefetch(arow + k0 + 64, 0, 1);          // global_prefetch_b8
#pragma unroll
        for (int bt = 0; bt < 4; ++bt) {
            // B fragment: lane = column n, 16 contiguous K values (one W row)
            const int ncol = ng * 64 + bt * 16 + ln;
            const __bf16* wrow = W + (size_t)ncol * K + k0 + lg * 16;
            FragBF b;
            b.u[0] = *(const u32x4*)(wrow);
            b.u[1] = *(const u32x4*)(wrow + 8);
            acc[bt] = __builtin_amdgcn_wmma_f32_16x16x32_bf16(
                false, a.v, false, b.v, (short)0, acc[bt], false, false);
        }
    }
#pragma unroll
    for (int bt = 0; bt < 4; ++bt) {
        const int ncol = ng * 64 + bt * 16 + ln;
        const float bv = bias[ncol];
#pragma unroll
        for (int r = 0; r < 8; ++r) {
            const int m = mt * 16 + r + 8 * lg;            // C: VGPR=row, lane=col
            storeOut(C + (size_t)m * N + ncol, acc[bt][r] + bv);
        }
    }
}

// ---------------------------------------------------------------------------
// Attention kernel: one block (7 waves) per (window b, head h).
// Wave qt owns q rows [qt*16, qt*16+16).  S = Q*K^T (7 WMMAs, K=HD=32),
// masked softmax in registers, P->LDS bf16, O = P*V (4 K-chunks x 2 d-tiles).
// All masking is branch-free (clamped loads + selects) so the compiler can
// clause the bias/mask loads instead of emitting saveexec chains.
// ---------------------------------------------------------------------------
__global__ __launch_bounds__(224)
void attn_kernel(const __bf16* __restrict__ qkv,      // [BW, NTOK, 1536] bf16
                 const float*  __restrict__ biasPre,  // [HEADS, 98, 98]
                 const float*  __restrict__ mask,     // [NW, 98, 98]
                 __bf16* __restrict__ out) {          // [BW, NTOK, 512] bf16
    __shared__ __bf16 vT[HD][128];        // V transposed, n padded to 128 w/ 0
    __shared__ __bf16 pP[7][16][128];     // softmax probs per q-tile

    const int b  = blockIdx.x >> 4;
    const int h  = blockIdx.x & 15;
    const int w  = b & (NW - 1);          // mask window index  (b % 64)
    const int tid  = threadIdx.x;
    const int lane = tid & 31;
    const int qt   = tid >> 5;            // wave id == q tile (0..6)
    const int ln   = lane & 15;
    const int lg   = lane >> 4;

    const __bf16* base = qkv + (size_t)b * NTOK * DIM3 + (size_t)h * HD;

    // ---- stage V transposed into LDS (zero pad n = 98..127), branch-free ----
    {
        const unsigned short* bs = (const unsigned short*)base;
        for (int idx = tid; idx < HD * 128; idx += 224) {
            const int d = idx >> 7;           // 0..31
            const int n = idx & 127;          // 0..127
            const int ncl = n < NTOK ? n : NTOK - 1;
            const unsigned short v = bs[(size_t)ncl * DIM3 + 1024 + d];
            *(unsigned short*)&vT[d][n] = (n < NTOK) ? v : (unsigned short)0;
        }
    }
    __syncthreads();

    // ---- Q fragment (A layout: lane = row, split K chunks) ----
    const int mq  = qt * 16 + ln;
    const int mqc = mq < NTOK ? mq : NTOK - 1;      // clamp; masked later
    FragBF qa;
    {
        const __bf16* qp = base + (size_t)mqc * DIM3 + lg * 16;
        qa.u[0] = *(const u32x4*)(qp);
        qa.u[1] = *(const u32x4*)(qp + 8);
    }

    // ---- S = Q * K^T : one WMMA per 16x16 tile (K = HD = 32) ----
    v8f s[7];
#pragma unroll
    for (int nt = 0; nt < 7; ++nt) {
        const int n  = nt * 16 + ln;
        const int nc = n < NTOK ? n : NTOK - 1;
        const __bf16* kp = base + (size_t)nc * DIM3 + 512 + lg * 16;
        FragBF kb;
        kb.u[0] = *(const u32x4*)(kp);
        kb.u[1] = *(const u32x4*)(kp + 8);
        v8f z = {};
        s[nt] = __builtin_amdgcn_wmma_f32_16x16x32_bf16(
            false, qa.v, false, kb.v, (short)0, s[nt] = z, false, false);
    }

    // ---- scale + bias + mask + range masking (branch-free) ----
    const float NEG = -1e30f;
    const float* bp = biasPre + (size_t)h * NN;
    const float* mp = mask    + (size_t)w * NN;
#pragma unroll
    for (int nt = 0; nt < 7; ++nt) {
        const int n   = nt * 16 + ln;
        const int ncl = n < NTOK ? n : NTOK - 1;
#pragma unroll
        for (int r = 0; r < 8; ++r) {
            const int m   = qt * 16 + r + 8 * lg;
            const int mcl = m < NTOK ? m : NTOK - 1;
            const int ix  = mcl * NTOK + ncl;
            const float add = bp[ix] + mp[ix];             // always in-bounds
            const float val = s[nt][r] * ATT_SCALE + add;
            s[nt][r] = ((m < NTOK) & (n < NTOK)) ? val : NEG;
        }
    }

    // ---- row softmax: each row lives in one VGPR across a 16-lane half ----
#pragma unroll
    for (int r = 0; r < 8; ++r) {
        float mx = NEG;
#pragma unroll
        for (int nt = 0; nt < 7; ++nt) mx = fmaxf(mx, s[nt][r]);
#pragma unroll
        for (int o = 1; o < 16; o <<= 1) mx = fmaxf(mx, __shfl_xor(mx, o, 32));
        float sum = 0.0f;
#pragma unroll
        for (int nt = 0; nt < 7; ++nt) {
            const float e = __expf(s[nt][r] - mx);
            s[nt][r] = e;
            sum += e;
        }
#pragma unroll
        for (int o = 1; o < 16; o <<= 1) sum += __shfl_xor(sum, o, 32);
        const float inv = 1.0f / sum;
        const int m    = qt * 16 + r + 8 * lg;
        const int mloc = r + 8 * lg;
#pragma unroll
        for (int nt = 0; nt < 7; ++nt) {
            const float p = (m < NTOK) ? s[nt][r] * inv : 0.0f;
            pP[qt][mloc][nt * 16 + ln] = f2bf(p);
        }
        pP[qt][mloc][112 + ln] = f2bf(0.0f);    // zero K pad 112..127
    }
    // pP/vT consumed by the same wave; same-wave DS ordering suffices.

    // ---- O = P * V : K padded to 128 -> 4 chunks x 2 d-tiles ----
    v8f o0 = {}, o1 = {};
#pragma unroll
    for (int c = 0; c < 4; ++c) {
        FragBF pa;                         // A layout from LDS
        const __bf16* pr = &pP[qt][ln][c * 32 + lg * 16];
        pa.u[0] = *(const u32x4*)(pr);
        pa.u[1] = *(const u32x4*)(pr + 8);
#pragma unroll
        for (int dt = 0; dt < 2; ++dt) {   // B layout: lane = d column
            const __bf16* vr = &vT[dt * 16 + ln][c * 32 + lg * 16];
            FragBF vb;
            vb.u[0] = *(const u32x4*)(vr);
            vb.u[1] = *(const u32x4*)(vr + 8);
            if (dt == 0)
                o0 = __builtin_amdgcn_wmma_f32_16x16x32_bf16(
                    false, pa.v, false, vb.v, (short)0, o0, false, false);
            else
                o1 = __builtin_amdgcn_wmma_f32_16x16x32_bf16(
                    false, pa.v, false, vb.v, (short)0, o1, false, false);
        }
    }

    // ---- store O as bf16 [b, m, h*32 + d] ----
#pragma unroll
    for (int dt = 0; dt < 2; ++dt) {
        const v8f o = dt ? o1 : o0;
#pragma unroll
        for (int r = 0; r < 8; ++r) {
            const int m = qt * 16 + r + 8 * lg;
            if (m < NTOK)
                out[((size_t)(b * NTOK + m)) * DIM + h * HD + dt * 16 + ln]
                    = f2bf(o[r]);
        }
    }
}

// ---------------------------------------------------------------------------
// Host launcher
// ---------------------------------------------------------------------------
extern "C" void kernel_launch(void* const* d_in, const int* in_sizes, int n_in,
                              void* d_out, int out_size, void* d_ws, size_t ws_size,
                              hipStream_t stream) {
    const float* x          = (const float*)d_in[0];
    const float* mask       = (const float*)d_in[1];
    const float* qkv_w      = (const float*)d_in[2];
    const float* qkv_b      = (const float*)d_in[3];
    const float* proj_w     = (const float*)d_in[4];
    const float* proj_b     = (const float*)d_in[5];
    const float* bias_table = (const float*)d_in[6];
    const int*   rel_index  = (const int*)d_in[7];
    float* out = (float*)d_out;

    const size_t Mrows = (size_t)BW * NTOK;        // 50176
    char* ws = (char*)d_ws;
    size_t off = 0;
    __bf16* Xbf   = (__bf16*)(ws + off); off += Mrows * DIM  * 2;  // also attnOut
    __bf16* Wqkv  = (__bf16*)(ws + off); off += (size_t)DIM3 * DIM * 2;
    __bf16* Wproj = (__bf16*)(ws + off); off += (size_t)DIM  * DIM * 2;
    __bf16* qkvb  = (__bf16*)(ws + off); off += Mrows * DIM3 * 2;
    float*  biasPre = (float*)(ws + off); off += (size_t)HEADS * NN * 4;
    (void)ws_size; (void)in_sizes; (void)n_in; (void)out_size;

    // 1) bf16 conversions
    const int nX = (int)(Mrows * DIM);
    f32_to_bf16_kernel<<<(nX + 255) / 256, 256, 0, stream>>>(x, Xbf, nX);
    f32_to_bf16_kernel<<<(DIM3 * DIM + 255) / 256, 256, 0, stream>>>(qkv_w, Wqkv, DIM3 * DIM);
    f32_to_bf16_kernel<<<(DIM * DIM + 255) / 256, 256, 0, stream>>>(proj_w, Wproj, DIM * DIM);

    // 2) relative position bias gather
    bias_gather_kernel<<<(NN * HEADS + 255) / 256, 256, 0, stream>>>(
        bias_table, rel_index, biasPre);

    // 3) QKV GEMM: [50176,512] x [1536,512]^T + b -> bf16 [50176,1536]
    {
        const int waves = ((int)Mrows / 16) * (DIM3 / 64);   // 75264
        gemm_bf16_wmma<__bf16><<<(waves + 7) / 8, 256, 0, stream>>>(
            Xbf, Wqkv, qkv_b, qkvb, (int)Mrows, DIM3, DIM);
    }

    // 4) attention (writes attnOut, aliasing Xbf which is now dead)
    attn_kernel<<<BW * HEADS, 224, 0, stream>>>(qkvb, biasPre, mask, Xbf);

    // 5) output projection: [50176,512] x [512,512]^T + b -> f32 d_out
    {
        const int waves = ((int)Mrows / 16) * (DIM / 64);    // 25088
        gemm_bf16_wmma<float><<<(waves + 7) / 8, 256, 0, stream>>>(
            Xbf, Wproj, proj_b, out, (int)Mrows, DIM, DIM);
    }
}
